// RK45Simulator_6794638262570
// MI455X (gfx1250) — compile-verified, matrix operand order checked by reading, not run on hardware
//
#include <hip/hip_runtime.h>
#include <math.h>

#define B_TOT 512
#define T_TOT 256
#define NU    64
#define NX    128
#define NH    1024
#define NZ    192   // NX + NU
#define NY    16
#define MT    16    // batch rows per workgroup
#define DTC   0.01f

typedef _Float16 v16h __attribute__((ext_vector_type(16)));
typedef _Float16 v8h  __attribute__((ext_vector_type(8)));
typedef float    v8f  __attribute__((ext_vector_type(8)));

// Fast tanh: native V_TANH_F32 on gfx1250, else branch-free exp2/rcp form.
// tanh(x) = 1 - 2/(exp2(2*log2(e)*x) + 1); limits: x->-inf gives -1, x->+inf gives +1.
__device__ __forceinline__ float fast_tanh(float x) {
#if __has_builtin(__builtin_amdgcn_tanhf)
    return __builtin_amdgcn_tanhf(x);
#else
    float t = __builtin_amdgcn_exp2f(x * 2.8853900817779268f);
    return 1.0f - 2.0f * __builtin_amdgcn_rcpf(t + 1.0f);
#endif
}

// Load one WMMA operand fragment (16 halves per lane) as two contiguous
// 16-byte chunks: K = [kb, kb+8) and [kb+16, kb+24) of one row.
// Caller passes p = row_base + k0 + kb.
__device__ __forceinline__ v16h frag_ld(const _Float16* __restrict__ p) {
    v8h lo = *(const v8h*)(p);
    v8h hi = *(const v8h*)(p + 16);
    v16h r;
#pragma unroll
    for (int i = 0; i < 8; ++i) { r[i] = lo[i]; r[i + 8] = hi[i]; }
    return r;
}

__global__ void convert_w(const float* __restrict__ W1, const float* __restrict__ W2,
                          const float* __restrict__ C,  const float* __restrict__ D,
                          _Float16* __restrict__ W1h, _Float16* __restrict__ W2h,
                          _Float16* __restrict__ CDh) {
    int i      = blockIdx.x * blockDim.x + threadIdx.x;
    int stride = gridDim.x * blockDim.x;
    for (int idx = i; idx < NH * NZ; idx += stride) W1h[idx] = (_Float16)W1[idx];
    for (int idx = i; idx < NX * NH; idx += stride) W2h[idx] = (_Float16)W2[idx];
    for (int idx = i; idx < NY * NZ; idx += stride) {
        int r = idx / NZ, c = idx % NZ;
        CDh[idx] = (_Float16)((c < NX) ? C[r * NX + c] : D[r * NU + (c - NX)]);
    }
}

__global__ __launch_bounds__(256, 1) void rk45_persist(
    const float* __restrict__ u,   const float* __restrict__ x0,
    const _Float16* __restrict__ W1h, const float* __restrict__ b1,
    const _Float16* __restrict__ W2h, const float* __restrict__ b2,
    const _Float16* __restrict__ CDh,
    float* __restrict__ Xout, float* __restrict__ Yout) {

    __shared__ __attribute__((aligned(16))) float     x_s[MT * NX];        // 8 KB
    __shared__ __attribute__((aligned(16))) float     k_s[6][MT * NX];     // 48 KB
    __shared__ __attribute__((aligned(16))) _Float16  z_s[MT * NZ];        // 6 KB
    __shared__ __attribute__((aligned(16))) _Float16  h_s[MT * NH];        // 32 KB

    const int tid  = threadIdx.x;
    const int lane = tid & 31;
    const int wv   = tid >> 5;            // 8 waves
    const int nl   = lane & 15;           // N (or M for A-frag) within tile
    const int kb   = (lane >> 4) * 8;     // K sub-offset for hi half-wave
    const int mb   = (lane >> 4) * 8;     // M base for C/D fragment rows
    const int bm0  = blockIdx.x * MT;

    // RK45 stage coefficients (dt folded in)
    const float AC[6][5] = {
        {0.f, 0.f, 0.f, 0.f, 0.f},
        {DTC * 0.25f, 0.f, 0.f, 0.f, 0.f},
        {DTC * 3.f / 32.f, DTC * 9.f / 32.f, 0.f, 0.f, 0.f},
        {DTC * 1932.f / 2197.f, DTC * -7200.f / 2197.f, DTC * 7296.f / 2197.f, 0.f, 0.f},
        {DTC * 439.f / 216.f, DTC * -8.f, DTC * 3680.f / 513.f, DTC * -845.f / 4104.f, 0.f},
        {DTC * -8.f / 27.f, DTC * 2.f, DTC * -3544.f / 2565.f, DTC * 1859.f / 4104.f, DTC * -11.f / 40.f}};
    const float BC0 = DTC * 16.f / 135.f;
    const float BC2 = DTC * 6656.f / 12825.f;
    const float BC3 = DTC * 28561.f / 56430.f;
    const float BC4 = DTC * -9.f / 50.f;
    const float BC5 = DTC * 2.f / 55.f;

    // init state
    for (int idx = tid; idx < MT * NX; idx += 256)
        x_s[idx] = x0[(size_t)(bm0 + (idx >> 7)) * NX + (idx & 127)];
    __syncthreads();

    for (int t = 0; t < T_TOT; ++t) {
        // emit X (pre-update state), stage-invariant u columns of z
        for (int idx = tid; idx < MT * NX; idx += 256)
            Xout[(size_t)(bm0 + (idx >> 7)) * (T_TOT * NX) + (size_t)t * NX + (idx & 127)] = x_s[idx];
        for (int idx = tid; idx < MT * NU; idx += 256) {
            int m = idx >> 6, j = idx & 63;
            z_s[m * NZ + NX + j] =
                (_Float16)u[(size_t)(bm0 + m) * (T_TOT * NU) + (size_t)t * NU + j];
        }
        __syncthreads();

#pragma unroll
        for (int s = 0; s < 6; ++s) {
            // ---- build xs = x + sum_p AC[s][p] * k_p into z columns [0,128)
            for (int idx = tid; idx < MT * NX; idx += 256) {
                float v = x_s[idx];
#pragma unroll
                for (int p = 0; p < 5; ++p)
                    if (p < s) v += AC[s][p] * k_s[p][idx];
                int m = idx >> 7, j = idx & 127;
                z_s[m * NZ + j] = (_Float16)v;
            }
            __syncthreads();

            // ---- GEMM1: H = tanh(Z @ W1^T + b1), 64 N-tiles, K=192
            {
                const _Float16* ap = z_s + nl * NZ + kb;
#pragma unroll 2
                for (int tt = 0; tt < 8; ++tt) {
                    const int tile = wv * 8 + tt;
                    const int ng   = tile * 16 + nl;
                    const _Float16* bp = W1h + (size_t)ng * NZ + kb;
                    v8f acc = {};
#pragma unroll
                    for (int ks = 0; ks < 6; ++ks) {
                        v16h a = frag_ld(ap + ks * 32);
                        v16h b = frag_ld(bp + ks * 32);
                        acc = __builtin_amdgcn_wmma_f32_16x16x32_f16(
                            false, a, false, b, (short)0, acc, false, false);
                    }
                    const float bv = b1[ng];
#pragma unroll
                    for (int i = 0; i < 8; ++i)
                        h_s[(mb + i) * NH + ng] = (_Float16)fast_tanh(acc[i] + bv);
                }
                // y = [x;u] @ [C|D]^T : stage 1 only, wave 0 (uniform branch)
                if (s == 0 && wv == 0) {
                    const _Float16* bp = CDh + nl * NZ + kb;
                    v8f ya = {};
#pragma unroll
                    for (int ks = 0; ks < 6; ++ks) {
                        v16h a = frag_ld(ap + ks * 32);
                        v16h b = frag_ld(bp + ks * 32);
                        ya = __builtin_amdgcn_wmma_f32_16x16x32_f16(
                            false, a, false, b, (short)0, ya, false, false);
                    }
#pragma unroll
                    for (int i = 0; i < 8; ++i)
                        Yout[(size_t)(bm0 + mb + i) * (T_TOT * NY) + (size_t)t * NY + nl] = ya[i];
                }
            }
            __syncthreads();

            // ---- GEMM2: k_s = H @ W2^T + b2, 8 N-tiles (one per wave), K=1024
            {
                const int ng = wv * 16 + nl;
                const _Float16* ap = h_s + nl * NH + kb;
                const _Float16* bp = W2h + (size_t)ng * NH + kb;
                v8f acc = {};
#pragma unroll 4
                for (int ks = 0; ks < 32; ++ks) {
                    v16h a = frag_ld(ap + ks * 32);
                    v16h b = frag_ld(bp + ks * 32);
                    acc = __builtin_amdgcn_wmma_f32_16x16x32_f16(
                        false, a, false, b, (short)0, acc, false, false);
                }
                const float bv = b2[ng];
#pragma unroll
                for (int i = 0; i < 8; ++i)
                    k_s[s][(mb + i) * NX + ng] = acc[i] + bv;
            }
            __syncthreads();
        }

        // ---- x += dt * (16/135 k1 + 6656/12825 k3 + 28561/56430 k4 - 9/50 k5 + 2/55 k6)
        for (int idx = tid; idx < MT * NX; idx += 256) {
            x_s[idx] += BC0 * k_s[0][idx] + BC2 * k_s[2][idx] + BC3 * k_s[3][idx] +
                        BC4 * k_s[4][idx] + BC5 * k_s[5][idx];
        }
        __syncthreads();
    }
}

extern "C" void kernel_launch(void* const* d_in, const int* in_sizes, int n_in,
                              void* d_out, int out_size, void* d_ws, size_t ws_size,
                              hipStream_t stream) {
    const float* u  = (const float*)d_in[0];
    const float* x0 = (const float*)d_in[1];
    const float* W1 = (const float*)d_in[2];
    const float* b1 = (const float*)d_in[3];
    const float* W2 = (const float*)d_in[4];
    const float* b2 = (const float*)d_in[5];
    const float* C  = (const float*)d_in[6];
    const float* D  = (const float*)d_in[7];

    char* ws = (char*)d_ws;
    _Float16* W1h = (_Float16*)ws;                                               // 1024*192 f16
    _Float16* W2h = (_Float16*)(ws + (size_t)NH * NZ * 2);                       // 128*1024 f16
    _Float16* CDh = (_Float16*)(ws + (size_t)NH * NZ * 2 + (size_t)NX * NH * 2); // 16*192 f16

    float* Xout = (float*)d_out;
    float* Yout = Xout + (size_t)B_TOT * T_TOT * NX;

    convert_w<<<384, 256, 0, stream>>>(W1, W2, C, D, W1h, W2h, CDh);
    rk45_persist<<<B_TOT / MT, 256, 0, stream>>>(u, x0, W1h, b1, W2h, b2, CDh, Xout, Yout);
}
